// VectorQuantizer_19456201850957
// MI455X (gfx1250) — matrix operand order repaired and checked by reading, not compiled, not run on hardware
//
#include <hip/hip_runtime.h>

typedef __attribute__((ext_vector_type(16))) _Float16 v16h;
typedef __attribute__((ext_vector_type(8)))  float    v8f;
typedef __attribute__((ext_vector_type(4)))  unsigned int v4u;
typedef __attribute__((ext_vector_type(4)))  unsigned int u32x4;
typedef __attribute__((ext_vector_type(8)))  int          i32x8;
typedef __attribute__((ext_vector_type(4)))  int          i32x4;

#define CCH     64        // channels (GEMM reduction depth)
#define KCODES  1024      // codebook size
#define NROWS   131072    // B*H*W
#define NUMEL   8388608   // B*C*H*W
#define HWMASK  4095      // H*W - 1

// d_out layout (floats): [z_q_st | vq_loss | perplexity | indices]
#define LOSS_OFF 8388608
#define PERP_OFF 8388609
#define IDX_OFF  8388610

// d_ws layout: float region then f16 embedding copy
#define WS_COUNTS 0        // 1024 floats
#define WS_SQSUM  1024     // 1 float
#define WS_E2     2048     // 1024 floats
#define WS_EMBH_BYTES 16384 // _Float16 embh[1024*64] = 128 KB

union FragH { v16h v; v4u q[2]; };
union FragF { v8f  v; float f[8]; };

// ---------- prep: zero accumulators, f32->f16 codebook, ||e_k||^2 ----------
__global__ __launch_bounds__(256)
void vq_prep(const float* __restrict__ emb, float* __restrict__ wsf,
             _Float16* __restrict__ embh) {
  int k = blockIdx.x * 256 + threadIdx.x;   // grid = 4 -> k in [0,1024)
  if (k < KCODES) {
    wsf[WS_COUNTS + k] = 0.0f;
    float s = 0.0f;
    #pragma unroll 8
    for (int c = 0; c < CCH; ++c) {
      float v = emb[k * CCH + c];
      s += v * v;
      embh[k * CCH + c] = (_Float16)v;
    }
    wsf[WS_E2 + k] = s;
  }
  if (k == 0) wsf[WS_SQSUM] = 0.0f;
}

// ---------- main: TDM stage -> WMMA scores -> argmin -> gather/loss/hist ----------
__global__ __launch_bounds__(256)
void vq_main(const float* __restrict__ ze, const float* __restrict__ emb,
             const _Float16* __restrict__ embh, float* __restrict__ wsf,
             float* __restrict__ out) {
  // TDM target: f32 tile in [c][m] order (64 lines of 16 contiguous floats)
  __shared__ __align__(16) float    aLds32[CCH * 16];  // 4 KB
  __shared__ __align__(16) _Float16 aLdsH[16 * CCH];   // 2 KB, [row][c] f16
  __shared__ float e2Lds[KCODES];
  __shared__ float redV[8][16];
  __shared__ int   redI[8][16];
  __shared__ int   bestIdx[16];

  const int t    = threadIdx.x;
  const int wave = t >> 5;
  const int lane = t & 31;
  const int n0   = blockIdx.x * 16;     // first flat row of this block
  const int bb   = n0 >> 12;            // batch index (16 rows never cross batch)
  const int hw0  = n0 & HWMASK;         // h*W + w of first row

  // Stage ||e||^2 table with regular loads while the TDM runs.
  #pragma unroll
  for (int j = 0; j < 4; ++j) {
    int idx = t + 256 * j;
    e2Lds[idx] = wsf[WS_E2 + idx];
  }

  // --- Tensor Data Mover: z_e[b, 0:64, hw0:hw0+16] -> aLds32[c][m] ---
  // D# group0: count=1 | lds_addr | global_addr(57b) | type=2
  // D# group1: data_size=4B, tensor_dim0=16, tensor_dim1=64, tile 16x64,
  //            tensor_dim0_stride=4096 elements
  if (t < 32) {  // wave 0 only, EXEC all-ones within the wave
    unsigned long long gaddr =
        (unsigned long long)(const char*)(ze + ((size_t)(bb * CCH) << 12) + hw0);
    unsigned int lds_off = (unsigned int)(unsigned long long)(&aLds32[0]);
    u32x4 g0 = { 1u,                                  // count=1, load, user
                 lds_off,
                 (unsigned int)gaddr,
                 (unsigned int)((gaddr >> 32) & 0x1FFFFFFu) | 0x80000000u };
    i32x8 g1 = { 0x00020000,      // data_size=2 (4 bytes), no flags, mask=0
                 16 << 16,        // tensor_dim0[15:0]=16
                 64 << 16,        // tensor_dim0 hi=0 | tensor_dim1[15:0]=64
                 16 << 16,        // tensor_dim1 hi=0 | tile_dim0=16
                 64,              // tile_dim1=64 | tile_dim2=0
                 4096,            // tensor_dim0_stride lo
                 0,               // stride0 hi | stride1 lo
                 0 };             // stride1 hi
    i32x4 g2 = { 0, 0, 0, 0 };
    i32x4 g3 = { 0, 0, 0, 0 };
    i32x8 g4 = { 0, 0, 0, 0, 0, 0, 0, 0 };
    __builtin_amdgcn_tensor_load_to_lds(g0, g1, g2, g3, g4, 0);
    __builtin_amdgcn_s_wait_tensorcnt(0);
  }
  __syncthreads();

  // Transpose + convert f32 [c][m] -> f16 [row][c] for WMMA fragment loads.
  #pragma unroll
  for (int j = 0; j < 4; ++j) {
    int idx = t + 256 * j;            // 0..1023
    int m = idx & 15, c = idx >> 4;
    aLdsH[m * CCH + c] = (_Float16)aLds32[c * 16 + m];
  }
  __syncthreads();

  // A fragments per ISA 16-bit 16x32 layout:
  // lane row M = lane&15; half = lane>>4; i<8 -> K=half*8+i, i>=8 -> K=16+half*8+(i-8)
  const int M     = lane & 15;
  const int khalf = lane >> 4;
  const _Float16* arow = aLdsH + M * CCH;
  FragH a0, a1;
  a0.q[0] = *(const v4u*)(arow +      khalf * 8);
  a0.q[1] = *(const v4u*)(arow + 16 + khalf * 8);
  a1.q[0] = *(const v4u*)(arow + 32 + khalf * 8);
  a1.q[1] = *(const v4u*)(arow + 48 + khalf * 8);

  float minv[8];
  int   mini[8];
  #pragma unroll
  for (int r = 0; r < 8; ++r) { minv[r] = 3.4e38f; mini[r] = 0; }

  // Each wave handles 8 code tiles of 16 codes -> all 1024 codes covered.
  // Fully unrolled: lets the scheduler clause the global b128 loads ahead of
  // the WMMA chain instead of load->wait->wmma lockstep per iteration.
  #pragma unroll
  for (int j = 0; j < 8; ++j) {
    const int kc0  = (wave * 8 + j) * 16;
    const int code = kc0 + M;           // this lane's column (code id)
    // B 32x16 f16 layout: same K order as A: i<8 -> K=half*8+i, i>=8 -> K=16+half*8+(i-8)
    const _Float16* bp = embh + code * CCH + khalf * 8;
    FragH b0, b1;
    b0.q[0] = *(const v4u*)(bp);
    b0.q[1] = *(const v4u*)(bp + 16);
    b1.q[0] = *(const v4u*)(bp + 32);
    b1.q[1] = *(const v4u*)(bp + 48);

    v8f accv = {};
    accv = __builtin_amdgcn_wmma_f32_16x16x32_f16(false, a0.v, false, b0.v,
                                                  (short)0, accv, false, false);
    accv = __builtin_amdgcn_wmma_f32_16x16x32_f16(false, a1.v, false, b1.v,
                                                  (short)0, accv, false, false);
    FragF acc; acc.v = accv;

    // Per-lane codes strictly increase over j -> plain '<' keeps first occurrence.
    float e2c = e2Lds[code];
    #pragma unroll
    for (int r = 0; r < 8; ++r) {
      float d = e2c - 2.0f * acc.f[r];  // ||z||^2 constant per row: dropped
      if (d < minv[r]) { minv[r] = d; mini[r] = code; }
    }
  }

  // Cross-lane argmin over 16 columns per half (wave32 butterfly, tie -> low idx).
  #pragma unroll
  for (int m = 1; m <= 8; m <<= 1) {
    #pragma unroll
    for (int r = 0; r < 8; ++r) {
      float ov = __shfl_xor(minv[r], m, 32);
      int   oi = __shfl_xor(mini[r], m, 32);
      if (ov < minv[r] || (ov == minv[r] && oi < mini[r])) {
        minv[r] = ov; mini[r] = oi;
      }
    }
  }
  if ((lane & 15) == 0) {
    #pragma unroll
    for (int r = 0; r < 8; ++r) {
      redV[wave][khalf * 8 + r] = minv[r];
      redI[wave][khalf * 8 + r] = mini[r];
    }
  }
  __syncthreads();

  // Cross-wave argmin; write indices and histogram.
  if (t < 16) {
    float bv = redV[0][t]; int bi = redI[0][t];
    #pragma unroll
    for (int w = 1; w < 8; ++w) {
      float v = redV[w][t]; int i2 = redI[w][t];
      if (v < bv || (v == bv && i2 < bi)) { bv = v; bi = i2; }
    }
    bestIdx[t] = bi;
    out[IDX_OFF + n0 + t] = (float)bi;
    atomicAdd(&wsf[WS_COUNTS + bi], 1.0f);
  }
  __syncthreads();

  // z_q_st == z_q numerically; scatter to NCHW + accumulate (z_e - z_q)^2.
  // z_e read from the f32 TDM tile in LDS -- no second HBM pass over z_e.
  float lsum = 0.0f;
  #pragma unroll
  for (int j = 0; j < 4; ++j) {
    int idx = t + 256 * j;
    int row = idx >> 6, c = idx & 63;
    size_t g = ((size_t)(bb * CCH + c) << 12) + (size_t)(hw0 + row);
    float q = emb[bestIdx[row] * CCH + c];   // f32 codebook for output precision
    out[g] = q;
    float d = aLds32[c * 16 + row] - q;
    lsum += d * d;
  }
  #pragma unroll
  for (int m = 16; m >= 1; m >>= 1) lsum += __shfl_xor(lsum, m, 32);
  if (lane == 0) atomicAdd(&wsf[WS_SQSUM], lsum);
}

// ---------- finalize: perplexity + vq_loss scalars ----------
__global__ __launch_bounds__(256)
void vq_final(const float* __restrict__ wsf, float* __restrict__ out) {
  __shared__ float red[8];
  int t = threadIdx.x;
  float acc = 0.0f;
  #pragma unroll
  for (int j = 0; j < 4; ++j) {
    int k = t + 256 * j;
    float p = wsf[WS_COUNTS + k] * (1.0f / (float)NROWS);
    acc += p * __logf(p + 1e-10f);
  }
  #pragma unroll
  for (int m = 16; m >= 1; m >>= 1) acc += __shfl_xor(acc, m, 32);
  if ((t & 31) == 0) red[t >> 5] = acc;
  __syncthreads();
  if (t == 0) {
    float h = 0.0f;
    #pragma unroll
    for (int w = 0; w < 8; ++w) h += red[w];
    out[PERP_OFF] = __expf(-h);
    out[LOSS_OFF] = (0.25f + 1.0f) * wsf[WS_SQSUM] * (1.0f / (float)NUMEL);
  }
}

extern "C" void kernel_launch(void* const* d_in, const int* in_sizes, int n_in,
                              void* d_out, int out_size, void* d_ws, size_t ws_size,
                              hipStream_t stream) {
  const float* ze  = (const float*)d_in[0];   // [32,64,64,64]
  const float* emb = (const float*)d_in[1];   // [1024,64]
  float*     wsf   = (float*)d_ws;
  _Float16*  embh  = (_Float16*)((char*)d_ws + WS_EMBH_BYTES);
  float*     out   = (float*)d_out;

  vq_prep <<<KCODES / 256, 256, 0, stream>>>(emb, wsf, embh);
  vq_main <<<NROWS / 16,   256, 0, stream>>>(ze, emb, embh, wsf, out);
  vq_final<<<1,            256, 0, stream>>>(wsf, out);
}
